// ObjectDetector1D_31748398252293
// MI455X (gfx1250) — matrix operand order repaired and checked by reading, not compiled, not run on hardware
//
#include <hip/hip_runtime.h>
#include <hip/hip_bf16.h>

typedef __attribute__((ext_vector_type(16))) __bf16 v16bf;
typedef __attribute__((ext_vector_type(8)))  __bf16 v8bf;
typedef __attribute__((ext_vector_type(8)))  float  v8f;

#define B_SZ    8
#define C_CH    256
#define NCLS    20
#define NCONV   4
#define LTOT    3584
#define A_TOT   7168
#define K_KEEP  1000
#define LTILE   128
#define HALO    5
#define SPAN    138     // LTILE + 2*HALO
#define NPAD    144     // 9 * 16 columns computed
#define NTT     9       // N tiles of 16
#define ROWS    146     // NPAD + 2 guard rows
#define CHP     264     // padded channel stride (bank-conflict relief)
#define KTOT    768     // Cin * 3 taps
#define KBLK    24      // KTOT / 32

// repacked weight regions (element offsets, bf16)
#define WB_CLS   0
#define WB_REG   786432          // 4*256*768
#define WB_CLSP  1572864         // + 4*256*768
#define WB_REGP  1609728         // + 48*768
#define WB_TOTAL 1622016         // + 16*768

static __device__ __forceinline__ __bf16 f2bf(float f) {
  unsigned u = __builtin_bit_cast(unsigned, f);
  unsigned r = (u + 0x7FFFu + ((u >> 16) & 1u)) >> 16;   // RNE
  return __builtin_bit_cast(__bf16, (unsigned short)r);
}
#define BF0 (__builtin_bit_cast(__bf16, (unsigned short)0))

static __device__ __forceinline__ v16bf load16(const __bf16* p0, const __bf16* p1) {
  v8bf lo = *(const v8bf*)p0;
  v8bf hi = *(const v8bf*)p1;
  v16bf r;
#pragma unroll
  for (int i = 0; i < 8; ++i) { r[i] = lo[i]; r[i + 8] = hi[i]; }
  return r;
}

// ---------------------------------------------------------------------------
// Kernel 0: repack fp32 [Cout][Cin][3] weights -> bf16 [Cout][t][Cin]
// ---------------------------------------------------------------------------
__global__ __launch_bounds__(256) void repack_kernel(
    const float* __restrict__ cls_w, const float* __restrict__ reg_w,
    const float* __restrict__ clsp_w, const float* __restrict__ regp_w,
    __bf16* __restrict__ wb)
{
  int i = blockIdx.x * 256 + threadIdx.x;
  if (i >= WB_TOTAL) return;
  float v;
  if (i < WB_REG) {                     // cls chain [4][256][256][3]
    int r = i;
    int layer = r / (C_CH * KTOT); int q = r % (C_CH * KTOT);
    int co = q / KTOT; int k = q % KTOT; int t = k / C_CH; int ci = k % C_CH;
    v = cls_w[((size_t)layer * C_CH + co) * KTOT + ci * 3 + t];
  } else if (i < WB_CLSP) {             // reg chain
    int r = i - WB_REG;
    int layer = r / (C_CH * KTOT); int q = r % (C_CH * KTOT);
    int co = q / KTOT; int k = q % KTOT; int t = k / C_CH; int ci = k % C_CH;
    v = reg_w[((size_t)layer * C_CH + co) * KTOT + ci * 3 + t];
  } else if (i < WB_REGP) {             // cls head [40][256][3] padded to 48 rows
    int r = i - WB_CLSP;
    int co = r / KTOT; int k = r % KTOT; int t = k / C_CH; int ci = k % C_CH;
    v = (co < 40) ? clsp_w[(size_t)co * KTOT + ci * 3 + t] : 0.f;
  } else {                              // reg head [4][256][3] padded to 16 rows
    int r = i - WB_REGP;
    int co = r / KTOT; int k = r % KTOT; int t = k / C_CH; int ci = k % C_CH;
    v = (co < 4) ? regp_w[(size_t)co * KTOT + ci * 3 + t] : 0.f;
  }
  wb[i] = f2bf(v);
}

// ---------------------------------------------------------------------------
// Kernel 1: fused 4-layer conv chain + head via implicit-GEMM WMMA (bf16)
// block = 256 threads (8 wave32); grid = (28 tiles, 8 batch, 2 subnets)
// Each wave owns a 32-row M strip; both 16-row M tiles live in registers so
// every B fragment read from LDS feeds TWO wmmas (halves LDS pressure).
// ---------------------------------------------------------------------------
__global__ __launch_bounds__(256) void conv_chain_kernel(
    const float* __restrict__ feat3, const float* __restrict__ feat4,
    const float* __restrict__ feat5,
    const __bf16* __restrict__ wb,
    const float* __restrict__ cls_b, const float* __restrict__ reg_b,
    const float* __restrict__ clsp_b, const float* __restrict__ regp_b,
    float* __restrict__ logits_ws, float* __restrict__ reg_ws)
{
  extern __shared__ char smem[];
  __bf16* bufA = (__bf16*)smem;
  __bf16* bufB = bufA + ROWS * CHP;
  float*  bias_lds = (float*)(smem + (size_t)2 * ROWS * CHP * sizeof(__bf16));

  const int tid  = threadIdx.x;
  const int lane = tid & 31;
  const int wid  = tid >> 5;
  const int tile = blockIdx.x;
  const int b    = blockIdx.y;
  const int sub  = blockIdx.z;

  int l0, L, lvl_off;
  const float* feat;
  if (tile < 16)      { l0 = tile * LTILE;        L = 2048; lvl_off = 0;    feat = feat3; }
  else if (tile < 24) { l0 = (tile - 16) * LTILE; L = 1024; lvl_off = 2048; feat = feat4; }
  else                { l0 = (tile - 24) * LTILE; L = 512;  lvl_off = 3072; feat = feat5; }

  const __bf16* chain_w = wb + (sub ? WB_REG  : WB_CLS);
  const float*  chain_b = sub ? reg_b  : cls_b;
  const __bf16* head_w  = wb + (sub ? WB_REGP : WB_CLSP);
  const float*  head_b  = sub ? regp_b : clsp_b;
  const int   mh_real   = sub ? 4 : 40;
  const int   mh_tiles  = sub ? 1 : 3;
  float*      head_out  = sub ? reg_ws : logits_ws;

  // zero both activation buffers (guard rows + pad columns stay zero)
  for (int i = tid; i < 2 * ROWS * CHP; i += 256) bufA[i] = BF0;
  __syncthreads();

  // stage input tile (coalesced along L), convert fp32 -> bf16
  const float* fb = feat + (size_t)b * C_CH * L;
  for (int i = tid; i < C_CH * SPAN; i += 256) {
    int ch = i / SPAN, p = i % SPAN;
    int g = l0 - HALO + p;
    float v = (g >= 0 && g < L) ? fb[(size_t)ch * L + g] : 0.0f;
    bufA[(p + 1) * CHP + ch] = f2bf(v);
  }
  __syncthreads();

  __bf16* src = bufA;
  __bf16* dst = bufB;

  // lane-constant fragment coordinates (ISA 7.12.2 layouts)
  const int abase = (lane < 16) ? 0 : 8;    // A: K-run base
  const int khalf = (lane < 16) ? 0 : 8;    // B: K-run base
  const int col   = lane & 15;              // B/C/D: column (N)
  const int coff  = (lane < 16) ? 0 : 8;    // C/D: row offset

  for (int layer = 0; layer < NCONV; ++layer) {
    for (int i = tid; i < C_CH; i += 256) bias_lds[i] = chain_b[layer * C_CH + i];
    __syncthreads();
    const __bf16* wl = chain_w + (size_t)layer * (C_CH * KTOT);

    const int m0   = wid * 32;               // this wave's 32-row M strip
    const int row0 = m0 + (lane & 15);       // A row for M tile 0 (tile1 = +16)
    v8f z = {};
    v8f acc0[NTT], acc1[NTT];
#pragma unroll
    for (int nt = 0; nt < NTT; ++nt) { acc0[nt] = z; acc1[nt] = z; }

    for (int kb = 0; kb < KBLK; ++kb) {
      const int tap = kb >> 3;
      const int cib = (kb & 7) * 32;
      const __bf16* ap0 = wl + (size_t)row0 * KTOT + kb * 32 + abase;
      const __bf16* ap1 = ap0 + 16 * KTOT;
      v16bf a0 = load16(ap0, ap0 + 16);
      v16bf a1 = load16(ap1, ap1 + 16);
      __builtin_prefetch(ap0 + 32, 0, 3);    // next K block toward WGP
#pragma unroll
      for (int nt = 0; nt < NTT; ++nt) {
        const __bf16* bp = src + (nt * 16 + col + tap) * CHP + cib + khalf;
        v16bf bbf = load16(bp, bp + 16);     // one B fragment ...
        acc0[nt] = __builtin_amdgcn_wmma_f32_16x16x32_bf16(
            false, a0, false, bbf, (short)0, acc0[nt], false, false);
        acc1[nt] = __builtin_amdgcn_wmma_f32_16x16x32_bf16(
            false, a1, false, bbf, (short)0, acc1[nt], false, false);  // ... two wmmas
      }
    }
    // epilogue: bias + relu, pack bf16, store 16B to LDS (both M tiles)
#pragma unroll
    for (int nt = 0; nt < NTT; ++nt) {
      const int pos = nt * 16 + col;
      const int ch0 = m0 + coff;
      v8bf rv0, rv1;
#pragma unroll
      for (int v = 0; v < 8; ++v) {
        float x0 = acc0[nt][v] + bias_lds[ch0 + v];
        float x1 = acc1[nt][v] + bias_lds[ch0 + 16 + v];
        rv0[v] = f2bf(x0 > 0.f ? x0 : 0.f);
        rv1[v] = f2bf(x1 > 0.f ? x1 : 0.f);
      }
      *(v8bf*)(dst + (pos + 1) * CHP + ch0)      = rv0;
      *(v8bf*)(dst + (pos + 1) * CHP + ch0 + 16) = rv1;
    }
    __syncthreads();
    // re-zero rows mapping outside [0, L) so next layer sees zero padding
    {
      int nb_front = (l0 == 0) ? HALO : 0;
      int pt = L - l0 + HALO; if (pt > NPAD) pt = NPAD;
      int nb_back = NPAD - pt;
      int nbad = nb_front + nb_back;
      for (int i = tid; i < nbad * C_CH; i += 256) {
        int rsel = i / C_CH, ch = i % C_CH;
        int p = (rsel < nb_front) ? rsel : pt + (rsel - nb_front);
        dst[(p + 1) * CHP + ch] = BF0;
      }
    }
    __syncthreads();
    { __bf16* t = dst; dst = src; src = t; }
  }

  // head conv (no relu), write fp32 to workspace
  const int pairs = mh_tiles * NTT;
  for (int pi = wid; pi < pairs; pi += 8) {
    const int mt = pi / NTT, nt = pi % NTT;
    const int m0 = mt * 16;
    const int row = m0 + (lane & 15);
    v8f acc = {};
    for (int kb = 0; kb < KBLK; ++kb) {
      const int tap = kb >> 3;
      const int cib = (kb & 7) * 32;
      const __bf16* ap = head_w + (size_t)row * KTOT + kb * 32 + abase;
      v16bf a = load16(ap, ap + 16);
      const __bf16* bp = src + (nt * 16 + col + tap) * CHP + cib + khalf;
      v16bf bbf = load16(bp, bp + 16);
      acc = __builtin_amdgcn_wmma_f32_16x16x32_bf16(
          false, a, false, bbf, (short)0, acc, false, false);
    }
    const int pos = nt * 16 + col;
    const int gl  = l0 + pos - HALO;
    if (pos >= HALO && pos < HALO + LTILE && gl < L) {
#pragma unroll
      for (int v = 0; v < 8; ++v) {
        int ch = m0 + coff + v;
        if (ch < mh_real)
          head_out[((size_t)b * mh_real + ch) * LTOT + lvl_off + gl] = acc[v] + head_b[ch];
      }
    }
  }
}

// ---------------------------------------------------------------------------
// Kernel 2: per-anchor max-class score (sigmoid) + anchor decode
// ---------------------------------------------------------------------------
__global__ __launch_bounds__(256) void decode_kernel(
    const float* __restrict__ logits_ws, const float* __restrict__ reg_ws,
    float* __restrict__ scores, float* __restrict__ segs)
{
  int gid = blockIdx.x * 256 + threadIdx.x;
  if (gid >= B_SZ * A_TOT) return;
  int b = gid / A_TOT, a = gid % A_TOT;

  int base, L, stride, lvl_off; float len0, len1;
  if (a < 4096)      { base = 0;    L = 2048; stride = 8;  lvl_off = 0;    len0 = 16.f;  len1 = 32.f;  }
  else if (a < 6144) { base = 4096; L = 1024; stride = 16; lvl_off = 2048; len0 = 64.f;  len1 = 128.f; }
  else               { base = 6144; L = 512;  stride = 32; lvl_off = 3072; len0 = 256.f; len1 = 512.f; }
  int al = a - base, j = al / L, l = al % L;

  float m = -1e30f;
#pragma unroll
  for (int nc = 0; nc < NCLS; ++nc) {
    float x = logits_ws[((size_t)b * 40 + j * NCLS + nc) * LTOT + lvl_off + l];
    m = fmaxf(m, x);
  }
  float score = 1.0f / (1.0f + __expf(-m));     // sigmoid monotone: max then sigmoid

  float d0 = reg_ws[((size_t)b * 4 + j * 2 + 0) * LTOT + lvl_off + l];
  float d1 = reg_ws[((size_t)b * 4 + j * 2 + 1) * LTOT + lvl_off + l];
  float c = (l + 0.5f) * (float)stride;
  float len = j ? len1 : len0;
  float a0 = rintf(c - len * 0.5f);             // jnp.round = RNE
  float a1 = rintf(c + len * 0.5f);

  scores[(size_t)b * A_TOT + a] = score;
  segs[((size_t)b * A_TOT + a) * 2 + 0] = a0 + d0;
  segs[((size_t)b * A_TOT + a) * 2 + 1] = a1 + d1;
}

// ---------------------------------------------------------------------------
// Kernel 3: per-batch top-1000 via bitonic sort of 8192 (padded) in LDS
// ---------------------------------------------------------------------------
__global__ __launch_bounds__(1024) void topk_kernel(
    const float* __restrict__ scores, const float* __restrict__ segs,
    float* __restrict__ top_s, float* __restrict__ segs_k)
{
  extern __shared__ char sm[];
  float* key = (float*)sm;
  int*   idx = (int*)(sm + 8192 * sizeof(float));
  const int b = blockIdx.x, tid = threadIdx.x;
  const int N = 8192;

  for (int i = tid; i < N; i += 1024) {
    key[i] = (i < A_TOT) ? scores[(size_t)b * A_TOT + i] : -1e30f;
    idx[i] = i;
  }
  __syncthreads();

  for (int k = 2; k <= N; k <<= 1) {
    for (int j = k >> 1; j > 0; j >>= 1) {
      for (int i = tid; i < N; i += 1024) {
        int ixj = i ^ j;
        if (ixj > i) {
          bool desc = ((i & k) == 0);
          float a = key[i], c = key[ixj];
          bool sw = desc ? (a < c) : (a > c);
          if (sw) {
            key[i] = c; key[ixj] = a;
            int t = idx[i]; idx[i] = idx[ixj]; idx[ixj] = t;
          }
        }
      }
      __syncthreads();
    }
  }
  for (int k = tid; k < K_KEEP; k += 1024) {
    top_s[(size_t)b * K_KEEP + k] = key[k];
    int id = idx[k];
    segs_k[((size_t)b * K_KEEP + k) * 2 + 0] = segs[((size_t)b * A_TOT + id) * 2 + 0];
    segs_k[((size_t)b * K_KEEP + k) * 2 + 1] = segs[((size_t)b * A_TOT + id) * 2 + 1];
  }
}

// ---------------------------------------------------------------------------
// Kernel 4: greedy 1D NMS (wave-synchronous, one block per batch) + emit out
// ---------------------------------------------------------------------------
__global__ __launch_bounds__(256) void nms_kernel(
    const float* __restrict__ top_s, const float* __restrict__ segs_k,
    float* __restrict__ out)
{
  __shared__ float s0[K_KEEP], s1[K_KEEP], sc[K_KEEP];
  __shared__ unsigned char sup[K_KEEP];
  const int b = blockIdx.x, tid = threadIdx.x;

  for (int k = tid; k < K_KEEP; k += 256) {
    s0[k] = segs_k[((size_t)b * K_KEEP + k) * 2 + 0];
    s1[k] = segs_k[((size_t)b * K_KEEP + k) * 2 + 1];
    sc[k] = top_s[(size_t)b * K_KEEP + k];
    sup[k] = 0;
  }
  __syncthreads();

  if (tid < 32) {                       // single wave, lockstep greedy NMS
    volatile unsigned char* vsup = sup;
    for (int i = 0; i < K_KEEP - 1; ++i) {
      if (vsup[i]) continue;            // uniform across lanes
      float x0 = s0[i], x1 = s1[i];
      float li = x1 - x0;
      for (int j = i + 1 + tid; j < K_KEEP; j += 32) {
        float mx = fmaxf(x0, s0[j]);
        float mn = fminf(x1, s1[j]);
        float inter = fmaxf(mn - mx, 0.0f);
        float uni = li + (s1[j] - s0[j]) - inter;
        float iou = inter / fmaxf(uni, 1e-6f);
        if (iou > 0.5f) vsup[j] = 1;
      }
    }
  }
  __syncthreads();

  for (int k = tid; k < K_KEEP; k += 256) {
    float keep = sup[k] ? 0.0f : 1.0f;
    out[((size_t)b * K_KEEP + k) * 3 + 0] = sc[k] * keep;
    out[((size_t)b * K_KEEP + k) * 3 + 1] = s0[k];
    out[((size_t)b * K_KEEP + k) * 3 + 2] = s1[k];
  }
}

// ---------------------------------------------------------------------------
extern "C" void kernel_launch(void* const* d_in, const int* in_sizes, int n_in,
                              void* d_out, int out_size, void* d_ws, size_t ws_size,
                              hipStream_t stream) {
  const float* feat3  = (const float*)d_in[0];
  const float* feat4  = (const float*)d_in[1];
  const float* feat5  = (const float*)d_in[2];
  const float* cls_w  = (const float*)d_in[3];
  const float* cls_b  = (const float*)d_in[4];
  const float* reg_w  = (const float*)d_in[5];
  const float* reg_b  = (const float*)d_in[6];
  const float* clsp_w = (const float*)d_in[7];
  const float* clsp_b = (const float*)d_in[8];
  const float* regp_w = (const float*)d_in[9];
  const float* regp_b = (const float*)d_in[10];

  char* ws = (char*)d_ws;
  size_t off = 0;
  auto alloc = [&](size_t bytes) -> void* {
    void* p = ws + off;
    off = (off + bytes + 255) & ~(size_t)255;
    return p;
  };
  __bf16* wb        = (__bf16*)alloc((size_t)WB_TOTAL * 2);
  float*  logits_ws = (float*)alloc((size_t)B_SZ * 40 * LTOT * 4);
  float*  reg_ws    = (float*)alloc((size_t)B_SZ * 4 * LTOT * 4);
  float*  scores    = (float*)alloc((size_t)B_SZ * A_TOT * 4);
  float*  segs      = (float*)alloc((size_t)B_SZ * A_TOT * 2 * 4);
  float*  top_s     = (float*)alloc((size_t)B_SZ * K_KEEP * 4);
  float*  segs_k    = (float*)alloc((size_t)B_SZ * K_KEEP * 2 * 4);

  // 0) weight repack fp32 -> bf16 K-major
  repack_kernel<<<(WB_TOTAL + 255) / 256, 256, 0, stream>>>(
      cls_w, reg_w, clsp_w, regp_w, wb);

  // 1) fused conv chains + heads (WMMA)
  size_t lds1 = (size_t)2 * ROWS * CHP * sizeof(__bf16) + C_CH * sizeof(float);
  conv_chain_kernel<<<dim3(28, B_SZ, 2), 256, lds1, stream>>>(
      feat3, feat4, feat5, wb, cls_b, reg_b, clsp_b, regp_b, logits_ws, reg_ws);

  // 2) scores + decoded segments
  decode_kernel<<<(B_SZ * A_TOT + 255) / 256, 256, 0, stream>>>(
      logits_ws, reg_ws, scores, segs);

  // 3) per-batch top-1000
  topk_kernel<<<B_SZ, 1024, 8192 * 8, stream>>>(scores, segs, top_s, segs_k);

  // 4) NMS + output
  nms_kernel<<<B_SZ, 256, 0, stream>>>(top_s, segs_k, (float*)d_out);
}
